// Net_57956288692302
// MI455X (gfx1250) — matrix operand order repaired and checked by reading, not compiled, not run on hardware
//
#include <hip/hip_runtime.h>

// ---------------------------------------------------------------------------
// MIP-GNN forward for MI455X (gfx1250, wave32, WMMA bf16).
// Dead code removed: the reference overwrites aggr[:, -1] after segment_sum,
// so the h2v MLP / edge_features path never affects the output.
// ---------------------------------------------------------------------------

#define N_VAR   25000
#define N_CON   25000
#define N_NODES 50000
#define N_EDGES 800000

typedef __attribute__((ext_vector_type(16))) __bf16 v16bf;
typedef __attribute__((ext_vector_type(8)))  float  v8f;

static __device__ inline v8f wmma_bf16(v16bf a, v16bf b, v8f c) {
  // D = A(16x32 bf16) x B(32x16 bf16) + C(16x16 f32)
  return __builtin_amdgcn_wmma_f32_16x16x32_bf16(
      /*neg_a=*/false, a, /*neg_b=*/false, b,
      /*c_mod=*/(short)0, c, /*reuse_a=*/false, /*reuse_b=*/false);
}

static __device__ inline unsigned short f2bf_bits(float f) {
  __bf16 h = (__bf16)f;
  return __builtin_bit_cast(unsigned short, h);
}

// ---------------------------------------------------------------------------
// Node embedding MLPs: x0_main[node][64], x0_last[node]
// thread group of 64 per node; thread j computes output column j.
// ---------------------------------------------------------------------------
__global__ __launch_bounds__(256) void embed_kernel(
    const float* __restrict__ varf, const float* __restrict__ conf,
    const float* __restrict__ vW1, const float* __restrict__ vB1,
    const float* __restrict__ vW2, const float* __restrict__ vB2,
    const float* __restrict__ cW1, const float* __restrict__ cB1,
    const float* __restrict__ cW2, const float* __restrict__ cB2,
    const int* __restrict__ assoc_var, const int* __restrict__ assoc_con,
    float* __restrict__ xmain, float* __restrict__ xlast) {
  int gid  = blockIdx.x * blockDim.x + threadIdx.x;
  int node = gid >> 6;
  int j    = gid & 63;
  if (node >= N_NODES) return;
  bool isVar = node < N_VAR;
  int  li    = isVar ? node : node - N_VAR;
  float v    = isVar ? varf[li] : conf[li];
  const float* W1 = isVar ? vW1 : cW1;
  const float* B1 = isVar ? vB1 : cB1;
  const float* W2 = isVar ? vW2 : cW2;
  const float* B2 = isVar ? vB2 : cB2;
  float acc = B2[j];
  for (int k = 0; k < 64; ++k) {
    float h = fmaxf(v * W1[k] + B1[k], 0.f);
    acc += h * W2[k * 64 + j];
  }
  int dn = isVar ? assoc_var[li] : assoc_con[li];
  xmain[(size_t)dn * 64 + j] = acc;
  if (j == 0) xlast[dn] = v;
}

// ---------------------------------------------------------------------------
// Per-conv weight build: w[r,K,N] = sum_b att[r,b]*basis[b,K,N] (K in 0..64).
// K<64 part is packed into the hardware WMMA B layout (bf16 bits):
//   flat = tile*512 + lane*16 + e ; tile = (rel*2+kt)*4+nt
//   K = kt*32 + (lane>>4)*8 + (e<8 ? e : e+8) ; N = nt*16 + (lane&15)
// K==64 row kept in f32 (rank-1 last-column term).
// ---------------------------------------------------------------------------
__global__ __launch_bounds__(256) void conv_wpack_kernel(
    const float* __restrict__ att, const float* __restrict__ basis,
    unsigned short* __restrict__ wpack, float* __restrict__ wlast) {
  int idx = blockIdx.x * blockDim.x + threadIdx.x;
  if (idx < 2 * 2 * 4 * 512) {
    int e    = idx & 15;
    int lane = (idx >> 4) & 31;
    int nt   = (idx >> 9) & 3;
    int kt   = (idx >> 11) & 1;
    int rel  = (idx >> 12) & 1;
    int K = kt * 32 + (lane >> 4) * 8 + (e < 8 ? e : e + 8);   // 0..63
    int N = nt * 16 + (lane & 15);            // 0..63
    float s = 0.f;
    for (int b = 0; b < 5; ++b)
      s += att[rel * 5 + b] * basis[((size_t)b * 65 + K) * 64 + N];
    wpack[idx] = f2bf_bits(s);
  } else if (idx < 8192 + 128) {
    int t   = idx - 8192;
    int rel = t >> 6;
    int N   = t & 63;
    float s = 0.f;
    for (int b = 0; b < 5; ++b)
      s += att[rel * 5 + b] * basis[((size_t)b * 65 + 64) * 64 + N];
    wlast[t] = s;
  }
}

// Generic fc B-pack: W is [Kact x 64] row-major f32, pad K to KT*32, bf16.
__global__ __launch_bounds__(256) void fc_pack_kernel(
    const float* __restrict__ W, int Kact, int KT,
    unsigned short* __restrict__ out) {
  int idx = blockIdx.x * blockDim.x + threadIdx.x;
  if (idx >= KT * 4 * 512) return;
  int e    = idx & 15;
  int lane = (idx >> 4) & 31;
  int nt   = (idx >> 9) & 3;
  int kt   = idx >> 11;
  int K = kt * 32 + (lane >> 4) * 8 + (e < 8 ? e : e + 8);
  int N = nt * 16 + (lane & 15);
  out[idx] = f2bf_bits(K < Kact ? W[(size_t)K * 64 + N] : 0.f);
}

// ---------------------------------------------------------------------------
// Edge kernel: one wave per 16-edge tile. Gather -> bf16 A tiles in registers
// (ISA 16-bit A layout), 2 relations x 2 K-tiles x 4 N-tiles of
// v_wmma_f32_16x16x32_bf16, per-edge relation select + rank-1 (K=64) term via
// __shfl, then f32 atomic scatter straight from the D layout (no LDS).
// ---------------------------------------------------------------------------
__global__ __launch_bounds__(256) void conv_edge_kernel(
    const float* __restrict__ xmain, const float* __restrict__ xlast,
    const int* __restrict__ src, const int* __restrict__ dst,
    const int* __restrict__ etype,
    const unsigned short* __restrict__ wpack, const float* __restrict__ wlast,
    float* __restrict__ amain) {
  int gtid = blockIdx.x * blockDim.x + threadIdx.x;
  int tile = gtid >> 5;
  int lane = threadIdx.x & 31;
  if (tile >= N_EDGES / 16) return;   // grid is exact: never taken
  int ebase = tile * 16;
  int m  = lane & 15;   // edge (M row) this lane serves
  int hi = lane >> 4;   // K-half selector

  int   sIdx = src[ebase + m];
  int   dIdx = dst[ebase + m];
  int   ty   = etype[ebase + m];
  float xl   = xlast[sIdx];

  // A tiles: lane<16 -> K {k0..k0+7, k0+16..k0+23}, lane>=16 -> +8 offset.
  const float4* r4 = (const float4*)(xmain + (size_t)sIdx * 64);
  v16bf A[2];
#pragma unroll
  for (int kt = 0; kt < 2; ++kt) {
    int k0 = kt * 32 + hi * 8;   // multiple of 8 -> float4 aligned
#pragma unroll
    for (int q = 0; q < 2; ++q) {
      float4 a = r4[(k0 >> 2) + q * 4];
      float4 b = r4[(k0 >> 2) + q * 4 + 1];
      A[kt][q * 8 + 0] = (__bf16)a.x;
      A[kt][q * 8 + 1] = (__bf16)a.y;
      A[kt][q * 8 + 2] = (__bf16)a.z;
      A[kt][q * 8 + 3] = (__bf16)a.w;
      A[kt][q * 8 + 4] = (__bf16)b.x;
      A[kt][q * 8 + 5] = (__bf16)b.y;
      A[kt][q * 8 + 6] = (__bf16)b.z;
      A[kt][q * 8 + 7] = (__bf16)b.w;
    }
  }

#pragma unroll
  for (int nt = 0; nt < 4; ++nt) {
    v8f acc0 = {0, 0, 0, 0, 0, 0, 0, 0};
    v8f acc1 = {0, 0, 0, 0, 0, 0, 0, 0};
#pragma unroll
    for (int kt = 0; kt < 2; ++kt) {
      v16bf b0 = *(const v16bf*)(wpack + ((0 * 2 + kt) * 4 + nt) * 512 + lane * 16);
      v16bf b1 = *(const v16bf*)(wpack + ((1 * 2 + kt) * 4 + nt) * 512 + lane * 16);
      acc0 = wmma_bf16(A[kt], b0, acc0);
      acc1 = wmma_bf16(A[kt], b1, acc1);
    }
    int   N   = nt * 16 + m;
    float wl0 = wlast[N];
    float wl1 = wlast[64 + N];
#pragma unroll
    for (int e = 0; e < 8; ++e) {
      int M = hi * 8 + e;                       // D layout row
      int   te = __shfl(ty, M);
      int   de = __shfl(dIdx, M);
      float xe = __shfl(xl, M);
      float v  = te ? acc1[e] : acc0[e];
      v += xe * (te ? wl1 : wl0);
      atomicAdd(amain + (size_t)de * 64 + N, v);
    }
  }
}

// ---------------------------------------------------------------------------
// Combine: x_out = relu(aggr + x@root + bias); last col = relu(x_last + ...).
// 64 threads/node, row staged in LDS (broadcast reads), root reads coalesced.
// ---------------------------------------------------------------------------
__global__ __launch_bounds__(256) void conv_combine_kernel(
    const float* __restrict__ amain,
    const float* __restrict__ xmain_in, const float* __restrict__ xlast_in,
    const float* __restrict__ root, const float* __restrict__ bias,
    float* __restrict__ xmain_out, float* __restrict__ xlast_out) {
  __shared__ float rowS[4][65];
  int slot = threadIdx.x >> 6;
  int j    = threadIdx.x & 63;
  int node = blockIdx.x * 4 + slot;
  bool ok  = node < N_NODES;
  rowS[slot][j] = ok ? xmain_in[(size_t)node * 64 + j] : 0.f;
  if (j == 0) rowS[slot][64] = ok ? xlast_in[node] : 0.f;
  __syncthreads();
  if (!ok) return;
  float acc = amain[(size_t)node * 64 + j] + bias[j];
  for (int k = 0; k < 65; ++k) acc += rowS[slot][k] * root[k * 65 + j];
  xmain_out[(size_t)node * 64 + j] = fmaxf(acc, 0.f);
  if (j == 0) {
    float a64 = rowS[slot][64] + bias[64];   // aggr[:, -1] := x[:, -1]
    for (int k = 0; k < 65; ++k) a64 += rowS[slot][k] * root[k * 65 + 64];
    xlast_out[node] = fmaxf(a64, 0.f);
  }
}

// ---------------------------------------------------------------------------
// Head: 16 var-rows per wave. fc1 (K=325 padded to 352) -> fc2 -> fc3 via
// WMMA with LDS D->A transposes, fc4 as a 64-wide dot per row.
// ---------------------------------------------------------------------------
__global__ __launch_bounds__(256) void head_kernel(
    const float* xm0, const float* xm1, const float* xm2, const float* xm3, const float* xm4,
    const float* xl0, const float* xl1, const float* xl2, const float* xl3, const float* xl4,
    const int* __restrict__ assoc_var,
    const unsigned short* __restrict__ B1, const float* __restrict__ fb1,
    const unsigned short* __restrict__ B2, const float* __restrict__ fb2,
    const unsigned short* __restrict__ B3, const float* __restrict__ fb3,
    const float* __restrict__ W4, const float* __restrict__ b4,
    float* __restrict__ out) {
  __shared__ float hbuf[8][16][64];   // per-wave private 16x64 staging
  int wv   = threadIdx.x >> 5;
  int lane = threadIdx.x & 31;
  int m  = lane & 15;
  int hi = lane >> 4;
  int rowbase = (blockIdx.x * 8 + wv) * 16;
  int row  = rowbase + m;
  bool rok = row < N_VAR;
  int node = rok ? assoc_var[row] : 0;
  const float* xms[5] = {xm0, xm1, xm2, xm3, xm4};
  const float* xls[5] = {xl0, xl1, xl2, xl3, xl4};

  v8f acc[4];
#pragma unroll
  for (int nt = 0; nt < 4; ++nt) acc[nt] = (v8f){0, 0, 0, 0, 0, 0, 0, 0};

  // ---- fc1: K = 325 (pad to 352 -> 11 K-tiles) ----
#pragma unroll
  for (int kt = 0; kt < 11; ++kt) {
    v16bf A;
#pragma unroll
    for (int e = 0; e < 16; ++e) {
      int K = kt * 32 + hi * 8 + (e < 8 ? e : e + 8);
      float v = 0.f;
      if (rok && K < 325) {
        int s = K / 65, off = K - s * 65;
        v = (off < 64) ? xms[s][(size_t)node * 64 + off] : xls[s][node];
      }
      A[e] = (__bf16)v;
    }
#pragma unroll
    for (int nt = 0; nt < 4; ++nt) {
      v16bf b = *(const v16bf*)(B1 + (kt * 4 + nt) * 512 + lane * 16);
      acc[nt] = wmma_bf16(A, b, acc[nt]);
    }
  }
#pragma unroll
  for (int nt = 0; nt < 4; ++nt) {
    int N = nt * 16 + m;
    float bb = fb1[N];
#pragma unroll
    for (int e = 0; e < 8; ++e)
      hbuf[wv][hi * 8 + e][N] = fmaxf(acc[nt][e] + bb, 0.f);
  }
  __syncthreads();

  // ---- fc2, fc3 ----
#pragma unroll
  for (int layer = 0; layer < 2; ++layer) {
    const unsigned short* Bx = (layer == 0) ? B2 : B3;
    const float*          bx = (layer == 0) ? fb2 : fb3;
#pragma unroll
    for (int nt = 0; nt < 4; ++nt) acc[nt] = (v8f){0, 0, 0, 0, 0, 0, 0, 0};
#pragma unroll
    for (int kt = 0; kt < 2; ++kt) {
      v16bf A;
#pragma unroll
      for (int e = 0; e < 16; ++e) {
        int K = kt * 32 + hi * 8 + (e < 8 ? e : e + 8);
        A[e] = (__bf16)hbuf[wv][m][K];
      }
#pragma unroll
      for (int nt = 0; nt < 4; ++nt) {
        v16bf b = *(const v16bf*)(Bx + (kt * 4 + nt) * 512 + lane * 16);
        acc[nt] = wmma_bf16(A, b, acc[nt]);
      }
    }
    __syncthreads();
#pragma unroll
    for (int nt = 0; nt < 4; ++nt) {
      int N = nt * 16 + m;
      float bb = bx[N];
#pragma unroll
      for (int e = 0; e < 8; ++e)
        hbuf[wv][hi * 8 + e][N] = fmaxf(acc[nt][e] + bb, 0.f);
    }
    __syncthreads();
  }

  // ---- fc4: per-row dot64 ----
  if (lane < 16 && rok) {
    float a = b4[0];
#pragma unroll
    for (int k = 0; k < 64; ++k) a += hbuf[wv][lane][k] * W4[k];
    out[row] = a;
  }
}

// ---------------------------------------------------------------------------
// Host orchestration.
// Input flattening assumption: top-level inputs in setup_inputs() insertion
// order; the nested `params` dict expands to its JAX-tree leaves (dict keys
// sorted, tuples/lists in order):
//   0 var_node_features, 1 con_node_features, 2 edge_features,
//   3..6   con_mlp (W1,b1,W2,b2)
//   7+8l.. convs[l]: att, basis, bias, h2v_w1, h2v_b1, h2v_w2, h2v_b2, root
//   39..46 fc1 W,b fc2 W,b fc3 W,b fc4 W,b
//   47..50 var_mlp (W1,b1,W2,b2)
//   51 edge_index, 52 edge_types, 53 assoc_var, 54 assoc_con, 55 node_types
// ---------------------------------------------------------------------------
extern "C" void kernel_launch(void* const* d_in, const int* in_sizes, int n_in,
                              void* d_out, int out_size, void* d_ws, size_t ws_size,
                              hipStream_t stream) {
  (void)in_sizes; (void)n_in; (void)out_size; (void)ws_size;

  const float* varf = (const float*)d_in[0];
  const float* conf = (const float*)d_in[1];
  // d_in[2] edge_features: dead (only feeds overwritten last column)
  const float* cW1 = (const float*)d_in[3];
  const float* cB1 = (const float*)d_in[4];
  const float* cW2 = (const float*)d_in[5];
  const float* cB2 = (const float*)d_in[6];
  const float* fc1W = (const float*)d_in[39];
  const float* fc1B = (const float*)d_in[40];
  const float* fc2W = (const float*)d_in[41];
  const float* fc2B = (const float*)d_in[42];
  const float* fc3W = (const float*)d_in[43];
  const float* fc3B = (const float*)d_in[44];
  const float* fc4W = (const float*)d_in[45];
  const float* fc4B = (const float*)d_in[46];
  const float* vW1 = (const float*)d_in[47];
  const float* vB1 = (const float*)d_in[48];
  const float* vW2 = (const float*)d_in[49];
  const float* vB2 = (const float*)d_in[50];
  const int* eidx  = (const int*)d_in[51];
  const int* src   = eidx;
  const int* dst   = eidx + N_EDGES;
  const int* etype = (const int*)d_in[52];
  const int* assoc_var = (const int*)d_in[53];
  const int* assoc_con = (const int*)d_in[54];
  float* out = (float*)d_out;

  // Workspace carve (256B-aligned)
  char* base = (char*)d_ws;
  size_t off = 0;
  auto carve = [&](size_t bytes) -> void* {
    void* p = base + off;
    off += (bytes + 255) & ~(size_t)255;
    return p;
  };
  float* xm[5];
  float* xl[5];
  for (int k = 0; k < 5; ++k) xm[k] = (float*)carve((size_t)N_NODES * 64 * 4);
  for (int k = 0; k < 5; ++k) xl[k] = (float*)carve((size_t)N_NODES * 4);
  float*          amain = (float*)carve((size_t)N_NODES * 64 * 4);
  unsigned short* wpack = (unsigned short*)carve(8192 * 2);
  float*          wlast = (float*)carve(128 * 4);
  unsigned short* B1 = (unsigned short*)carve(11 * 4 * 512 * 2);
  unsigned short* B2 = (unsigned short*)carve(2 * 4 * 512 * 2);
  unsigned short* B3 = (unsigned short*)carve(2 * 4 * 512 * 2);

  // x0 (scatter targets) must start at zero
  hipMemsetAsync(xm[0], 0, (size_t)N_NODES * 64 * 4, stream);
  hipMemsetAsync(xl[0], 0, (size_t)N_NODES * 4, stream);

  embed_kernel<<<(N_NODES * 64) / 256, 256, 0, stream>>>(
      varf, conf, vW1, vB1, vW2, vB2, cW1, cB1, cW2, cB2,
      assoc_var, assoc_con, xm[0], xl[0]);

  fc_pack_kernel<<<(11 * 4 * 512 + 255) / 256, 256, 0, stream>>>(fc1W, 325, 11, B1);
  fc_pack_kernel<<<(2 * 4 * 512 + 255) / 256, 256, 0, stream>>>(fc2W, 64, 2, B2);
  fc_pack_kernel<<<(2 * 4 * 512 + 255) / 256, 256, 0, stream>>>(fc3W, 64, 2, B3);

  for (int l = 0; l < 4; ++l) {
    const float* att   = (const float*)d_in[7 + 8 * l + 0];
    const float* basis = (const float*)d_in[7 + 8 * l + 1];
    const float* bias  = (const float*)d_in[7 + 8 * l + 2];
    const float* root  = (const float*)d_in[7 + 8 * l + 7];

    conv_wpack_kernel<<<(8320 + 255) / 256, 256, 0, stream>>>(att, basis, wpack, wlast);
    hipMemsetAsync(amain, 0, (size_t)N_NODES * 64 * 4, stream);
    conv_edge_kernel<<<(N_EDGES / 16) / 8, 256, 0, stream>>>(
        xm[l], xl[l], src, dst, etype, wpack, wlast, amain);
    conv_combine_kernel<<<(N_NODES + 3) / 4, 256, 0, stream>>>(
        amain, xm[l], xl[l], root, bias, xm[l + 1], xl[l + 1]);
  }

  head_kernel<<<(N_VAR + 127) / 128, 256, 0, stream>>>(
      xm[0], xm[1], xm[2], xm[3], xm[4],
      xl[0], xl[1], xl[2], xl[3], xl[4],
      assoc_var, B1, fc1B, B2, fc2B, B3, fc3B, fc4W, fc4B, out);
  (void)assoc_con;
}
